// PlugLSTM_11699490914522
// MI455X (gfx1250) — compile-verified, matrix operand order checked by reading, not run on hardware
//
#include <hip/hip_runtime.h>
#include <hip/hip_bf16.h>
#include <math.h>

// ---------------------------------------------------------------------------
// PlugLSTM on MI455X (gfx1250):
//  * all GEMMs on v_wmma_f32_16x16x32_bf16 (bf16 in, f32 accumulate)
//  * recurrent h-tile and decoder weight panel staged into LDS with the
//    Tensor Data Mover (tensor_load_to_lds + s_wait_tensorcnt)
//  * fused [x_t ; h_{t-1}] K=1024 gate GEMM per LSTM step, f32 cell math
// ---------------------------------------------------------------------------

typedef __attribute__((ext_vector_type(16))) __bf16 v16bf;
typedef __attribute__((ext_vector_type(8)))  __bf16 v8bf;
typedef __attribute__((ext_vector_type(8)))  float  v8f;
typedef __attribute__((ext_vector_type(4)))  unsigned int u32x4;
typedef __attribute__((ext_vector_type(8)))  int          i32x8;
typedef __attribute__((ext_vector_type(4)))  int          i32x4;

constexpr int Vv = 512;   // vocab
constexpr int Tt = 128;   // time steps
constexpr int Hh = 512;   // hidden
constexpr int Ll = 2;     // layers
constexpr int Bb = 512;   // batch
constexpr int G4H  = 4 * Hh;        // 2048 gate columns
constexpr int KC   = 2 * Hh;        // fused K ([x;h]) = 1024
constexpr int NT_G = G4H / 16;      // 128 n-tiles (gates)
constexpr int KT_G = KC / 32;       // 32 k-tiles
constexpr int NT_D = Vv / 16;       // 32 n-tiles (decoder)
constexpr int KT_D = Hh / 32;       // 16 k-tiles

static __device__ __forceinline__ unsigned short f2bf(float f) {
  unsigned int u = __float_as_uint(f);
  u += 0x7FFFu + ((u >> 16) & 1u);          // round-to-nearest-even
  return (unsigned short)(u >> 16);
}
static __device__ __forceinline__ v16bf cat8(v8bf lo, v8bf hi) {
  union { v8bf h[2]; v16bf v; } u;
  u.h[0] = lo; u.h[1] = hi; return u.v;
}
static __device__ __forceinline__ float sigf(float x) {
  return 1.0f / (1.0f + expf(-x));
}

// --- Tensor Data Mover: issue a load described by D# groups 0/1 -------------
static __device__ __forceinline__ void tdm_load(u32x4 g0, i32x8 g1) {
  i32x4 z4 = {0, 0, 0, 0};
#if __clang_major__ >= 23
  i32x8 z8 = {0, 0, 0, 0, 0, 0, 0, 0};
  __builtin_amdgcn_tensor_load_to_lds(g0, g1, z4, z4, z8, 0);
#else
  __builtin_amdgcn_tensor_load_to_lds(g0, g1, z4, z4, 0);
#endif
}

// Build D# group0: count=1, lds byte address, 57-bit global address, type=2.
static __device__ __forceinline__ u32x4 tdm_g0(unsigned lds_addr,
                                               unsigned long long ga) {
  u32x4 g0;
  g0[0] = 1u;                                  // count=1, user mode
  g0[1] = lds_addr;                            // [63:32] lds_addr
  g0[2] = (unsigned)(ga & 0xffffffffu);        // global_addr lo
  g0[3] = (unsigned)((ga >> 32) & 0x01ffffffu) // global_addr hi (57-bit)
        | (2u << 30);                          // type = 2 ("image")
  return g0;
}

// Build D# group1 for a 2D (or 1D) bf16 tile: data_size=1 (2 bytes).
static __device__ __forceinline__ i32x8 tdm_g1(unsigned tensor_d0,
                                               unsigned tensor_d1,
                                               unsigned tile_d0,
                                               unsigned tile_d1,
                                               unsigned d0_stride) {
  i32x8 g1;
  g1[0] = (int)(1u << 16);                               // data_size=2B
  g1[1] = (int)((tensor_d0 & 0xffffu) << 16);            // dim0[15:0] @ [63:48]
  g1[2] = (int)(((tensor_d0 >> 16) & 0xffffu)            // dim0[31:16]
        |       ((tensor_d1 & 0xffffu) << 16));          // dim1[15:0]
  g1[3] = (int)(((tensor_d1 >> 16) & 0xffffu)            // dim1[31:16]
        |       ((tile_d0 & 0xffffu) << 16));            // tile_dim0
  g1[4] = (int)(tile_d1 & 0xffffu);                      // tile_dim1 (0=1D)
  g1[5] = (int)d0_stride;                                // dim0 stride lo32
  g1[6] = 0;                                             // stride hi / dim1str
  g1[7] = 0;
  return g1;
}

// ---------------------------------------------------------------------------
// Weight repack: Wfrag[((l*KT_G + kt)*NT_G + nt)*32 + lane][16 halves]
// B-tile fragment convention (mirror of the documented 16-bit A layout):
//   lane<16 -> kb=0, lane>=16 -> kb=8 ; col n = nt*16 + (lane&15)
//   e in [0,8):  k = kt*32 + kb + e ;  e in [8,16): k = kt*32 + kb + 16 + e-8
// ---------------------------------------------------------------------------
__global__ void prep_wcat(const float* __restrict__ Wih,
                          const float* __restrict__ Whh,
                          unsigned short* __restrict__ Wfrag) {
  long i = (long)blockIdx.x * blockDim.x + threadIdx.x;
  long total = (long)Ll * KT_G * NT_G * 32 * 16;
  if (i >= total) return;
  int  e = i & 15;
  int  lane = (i >> 4) & 31;
  long r = i >> 9;
  int  nt = (int)(r % NT_G); r /= NT_G;
  int  kt = (int)(r % KT_G);
  int  l  = (int)(r / KT_G);
  int  kb = (lane < 16) ? 0 : 8;
  int  k  = kt * 32 + kb + ((e < 8) ? e : 16 + (e - 8));
  int  n  = nt * 16 + (lane & 15);
  float v = (k < Hh) ? Wih[((long)l * G4H + n) * Hh + k]
                     : Whh[((long)l * G4H + n) * Hh + (k - Hh)];
  Wfrag[i] = f2bf(v);
}

__global__ void prep_dec(const float* __restrict__ decW,
                         unsigned short* __restrict__ Dfrag) {
  long i = (long)blockIdx.x * blockDim.x + threadIdx.x;
  long total = (long)KT_D * NT_D * 32 * 16;
  if (i >= total) return;
  int  e = i & 15;
  int  lane = (i >> 4) & 31;
  long r = i >> 9;
  int  nt = (int)(r % NT_D);
  int  kt = (int)(r / NT_D);
  int  kb = (lane < 16) ? 0 : 8;
  int  k  = kt * 32 + kb + ((e < 8) ? e : 16 + (e - 8));
  int  n  = nt * 16 + (lane & 15);
  Dfrag[i] = f2bf(decW[(long)n * Hh + k]);
}

__global__ void prep_bias(const float* __restrict__ bih,
                          const float* __restrict__ bhh,
                          float* __restrict__ bsum) {
  int i = blockIdx.x * blockDim.x + threadIdx.x;
  if (i < Ll * G4H) bsum[i] = bih[i] + bhh[i];
}

__global__ void init_state(unsigned short* __restrict__ hA,
                           unsigned short* __restrict__ hB,
                           float* __restrict__ c) {
  long i = (long)blockIdx.x * blockDim.x + threadIdx.x;
  if (i >= (long)Bb * Hh) return;
  hA[i] = 0; hB[i] = 0; c[i] = 0.0f;
}

// inp[b,t,h] = bf16( emb[trg(b,t)][h] + y[b]*yW[h] + yb[h] ), trg shifted by 1.
__global__ void embed_kernel(const int* __restrict__ x, const float* __restrict__ y,
                             const float* __restrict__ emb, const float* __restrict__ yW,
                             const float* __restrict__ yb,
                             unsigned short* __restrict__ inp) {
  long i = (long)blockIdx.x * blockDim.x + threadIdx.x;
  if (i >= (long)Bb * Tt * Hh) return;
  int  h  = (int)(i & (Hh - 1));
  long bt = i >> 9;                 // /H
  int  t  = (int)(bt & (Tt - 1));
  int  b  = (int)(bt >> 7);         // /T
  int  tok = (t == 0) ? Vv : x[b * Tt + t - 1];
  float v = emb[(long)tok * Hh + h] + y[b] * yW[h] + yb[h];
  inp[i] = f2bf(v);
}

// ---------------------------------------------------------------------------
// One LSTM time step. Block = 512 threads (16 waves) owns a 32x64 (batch x h)
// slice. The 32x512 bf16 h_{t-1} tile (32 KB, reused by all 16 waves) is
// DMA'd to LDS by the TDM, then K = [x_t (global) ; h_{t-1} (LDS)] = 1024.
// Gate pre-activations staged in LDS (aliasing the dead h tile), then the
// i/f/g/o cell update. h-state ping-pongs; c updated in place.
// ---------------------------------------------------------------------------
__global__ __launch_bounds__(512) void lstm_step(
    const unsigned short* __restrict__ Xin,     // [B,T,H] bf16 layer input
    const unsigned short* __restrict__ Wfrag,   // layer weight fragments
    const float* __restrict__ bsum,             // [4H] folded biases
    const unsigned short* __restrict__ hprev,   // [B,H] bf16
    unsigned short* __restrict__ hnext,         // [B,H] bf16
    float* __restrict__ cstate,                 // [B,H] f32 (in place)
    unsigned short* __restrict__ Hout,          // [B,T,H] bf16 layer output
    int t) {
  __shared__ union {
    unsigned short ah[32][Hh];                  // staged h_{t-1} tile (32 KB)
    float          gl[4][32][65];               // gate staging (33 KB)
  } sm;

  const int b0 = blockIdx.x * 32;
  const int h0 = blockIdx.y * 64;
  const int wave = threadIdx.x >> 5;
  const int lane = threadIdx.x & 31;

  // --- TDM: pull h_{t-1}[b0..b0+31][0..511] (contiguous 32 KB) into LDS ----
  if (threadIdx.x < 32) {
    unsigned lds = (unsigned)(size_t)(&sm.ah[0][0]);
    unsigned long long ga =
        (unsigned long long)(size_t)(hprev + (size_t)b0 * Hh);
    // 1D tile: 16384 bf16 elements
    tdm_load(tdm_g0(lds, ga), tdm_g1(16384u, 1u, 16384u, 0u, 16384u));
    __builtin_amdgcn_s_wait_tensorcnt(0);
  }
  __syncthreads();

  const int m   = wave & 1;                     // m-tile (0/1)
  const int tc0 = (wave >> 1) * 2;              // first of 2 tile-columns
  const int g0i = tc0 >> 2,       n0t = tc0 & 3;
  const int g1i = (tc0 + 1) >> 2, n1t = (tc0 + 1) & 3;
  const int ntg0 = g0i * (Hh / 16) + (h0 >> 4) + n0t;   // global n-tile
  const int ntg1 = g1i * (Hh / 16) + (h0 >> 4) + n1t;

  const int mrow = m * 16 + (lane & 15);        // block-local A row
  const int kb   = (lane < 16) ? 0 : 8;
  const unsigned short* xrow = Xin + ((long)(b0 + mrow) * Tt + t) * Hh;

  v8f acc0 = {}, acc1 = {};

  // --- K part 1: x_t contribution (A from global) --------------------------
  for (int kk = 0; kk < Hh; kk += 32) {
    const unsigned short* ap = xrow + kk + kb;
    v16bf A = cat8(*(const v8bf*)ap, *(const v8bf*)(ap + 16));
    const int kt = kk >> 5;
    v16bf B0 = *(const v16bf*)(Wfrag + (((long)kt * NT_G + ntg0) * 32 + lane) * 16);
    v16bf B1 = *(const v16bf*)(Wfrag + (((long)kt * NT_G + ntg1) * 32 + lane) * 16);
    __builtin_prefetch(Wfrag + (((long)(kt + 1) * NT_G + ntg0) * 32 + lane) * 16, 0, 3);
    acc0 = __builtin_amdgcn_wmma_f32_16x16x32_bf16(false, A, false, B0, (short)0, acc0, false, false);
    acc1 = __builtin_amdgcn_wmma_f32_16x16x32_bf16(false, A, false, B1, (short)0, acc1, false, false);
  }
  // --- K part 2: h_{t-1} contribution (A from LDS via ds_load) -------------
  const unsigned short* hrl = &sm.ah[mrow][0];
  for (int kk = 0; kk < Hh; kk += 32) {
    const unsigned short* ap = hrl + kk + kb;
    v16bf A = cat8(*(const v8bf*)ap, *(const v8bf*)(ap + 16));
    const int kt = (kk >> 5) + (Hh / 32);
    v16bf B0 = *(const v16bf*)(Wfrag + (((long)kt * NT_G + ntg0) * 32 + lane) * 16);
    v16bf B1 = *(const v16bf*)(Wfrag + (((long)kt * NT_G + ntg1) * 32 + lane) * 16);
    acc0 = __builtin_amdgcn_wmma_f32_16x16x32_bf16(false, A, false, B0, (short)0, acc0, false, false);
    acc1 = __builtin_amdgcn_wmma_f32_16x16x32_bf16(false, A, false, B1, (short)0, acc1, false, false);
  }

  __syncthreads();                              // ah is dead; reuse as gl

  const int rb = m * 16 + ((lane >> 4) << 3);   // block-local row base (0..31)
  const int cc = lane & 15;
#pragma unroll
  for (int r = 0; r < 8; ++r) {
    sm.gl[g0i][rb + r][n0t * 16 + cc] = acc0[r];
    sm.gl[g1i][rb + r][n1t * 16 + cc] = acc1[r];
  }
  __syncthreads();

  for (int e = threadIdx.x; e < 32 * 64; e += 512) {
    int mr = e >> 6, nc = e & 63;
    int b = b0 + mr, hc = h0 + nc;
    float iv = sm.gl[0][mr][nc] + bsum[0 * Hh + hc];
    float fv = sm.gl[1][mr][nc] + bsum[1 * Hh + hc];
    float gv = sm.gl[2][mr][nc] + bsum[2 * Hh + hc];
    float ov = sm.gl[3][mr][nc] + bsum[3 * Hh + hc];
    long  ci = (long)b * Hh + hc;
    float cn = sigf(fv) * cstate[ci] + sigf(iv) * tanhf(gv);
    float hn = sigf(ov) * tanhf(cn);
    cstate[ci] = cn;
    unsigned short hb = f2bf(hn);
    hnext[ci] = hb;
    Hout[((long)b * Tt + t) * Hh + hc] = hb;
  }
}

// ---------------------------------------------------------------------------
// Decoder: logits[bt, v] = H2[bt,:] @ decW^T + decb.  M=65536 N=512 K=512.
// Block = 64x64 tile; its full B panel (16 kt x 4 ntiles = 64 KB of bf16
// fragments, reused by 4 m-tile waves) is DMA'd once into LDS by the TDM.
// ---------------------------------------------------------------------------
__global__ __launch_bounds__(256) void decode_gemm(
    const unsigned short* __restrict__ Hin,     // [B*T, H] bf16
    const unsigned short* __restrict__ Dfrag,
    const float* __restrict__ decb,
    float* __restrict__ out) {
  __shared__ unsigned short bsm[KT_D * 4 * 32 * 16];   // 64 KB B panel

  const int m0 = blockIdx.x * 64;
  const int n0 = blockIdx.y * 64;
  const int wave = threadIdx.x >> 5;
  const int lane = threadIdx.x & 31;

  // --- TDM: 2D tile, 16 rows (kt) x 2048 elems, row stride NT_D*512 --------
  if (threadIdx.x < 32) {
    unsigned lds = (unsigned)(size_t)(&bsm[0]);
    unsigned long long ga =
        (unsigned long long)(size_t)(Dfrag + (size_t)(n0 >> 4) * 512);
    tdm_load(tdm_g0(lds, ga),
             tdm_g1(/*tensor_d0=*/2048u, /*tensor_d1=*/16u,
                    /*tile_d0=*/2048u, /*tile_d1=*/16u,
                    /*d0_stride=*/(unsigned)(NT_D * 512)));
    __builtin_amdgcn_s_wait_tensorcnt(0);
  }
  __syncthreads();

  const int mt  = wave & 3;
  const int ntp = wave >> 2;
  const int nt0 = ntp * 2, nt1 = nt0 + 1;

  const int arow = m0 + mt * 16 + (lane & 15);
  const int kb   = (lane < 16) ? 0 : 8;
  const unsigned short* ar = Hin + (long)arow * Hh;

  v8f acc0 = {}, acc1 = {};
  for (int kk = 0; kk < Hh; kk += 32) {
    const unsigned short* ap = ar + kk + kb;
    v16bf A = cat8(*(const v8bf*)ap, *(const v8bf*)(ap + 16));
    const int kt = kk >> 5;
    const unsigned short* bp0 = bsm + ((kt * 4 + nt0) * 32 + lane) * 16;
    const unsigned short* bp1 = bsm + ((kt * 4 + nt1) * 32 + lane) * 16;
    v16bf B0 = cat8(*(const v8bf*)bp0, *(const v8bf*)(bp0 + 8));
    v16bf B1 = cat8(*(const v8bf*)bp1, *(const v8bf*)(bp1 + 8));
    acc0 = __builtin_amdgcn_wmma_f32_16x16x32_bf16(false, A, false, B0, (short)0, acc0, false, false);
    acc1 = __builtin_amdgcn_wmma_f32_16x16x32_bf16(false, A, false, B1, (short)0, acc1, false, false);
  }

  const int row  = m0 + mt * 16 + ((lane >> 4) << 3);
  const int col0 = n0 + nt0 * 16 + (lane & 15);
  const int col1 = n0 + nt1 * 16 + (lane & 15);
  const float b0v = decb[col0], b1v = decb[col1];
#pragma unroll
  for (int r = 0; r < 8; ++r) {
    out[(long)(row + r) * Vv + col0] = acc0[r] + b0v;
    out[(long)(row + r) * Vv + col1] = acc1[r] + b1v;
  }
}

// ---------------------------------------------------------------------------
extern "C" void kernel_launch(void* const* d_in, const int* in_sizes, int n_in,
                              void* d_out, int out_size, void* d_ws, size_t ws_size,
                              hipStream_t stream) {
  (void)in_sizes; (void)n_in; (void)out_size; (void)ws_size;
  const int*   x    = (const int*)  d_in[0];
  const float* y    = (const float*)d_in[1];
  const float* emb  = (const float*)d_in[2];
  const float* yW   = (const float*)d_in[3];
  const float* yb   = (const float*)d_in[4];
  const float* Wih  = (const float*)d_in[5];
  const float* Whh  = (const float*)d_in[6];
  const float* bih  = (const float*)d_in[7];
  const float* bhh  = (const float*)d_in[8];
  const float* decW = (const float*)d_in[9];
  const float* decb = (const float*)d_in[10];
  float* out = (float*)d_out;

  // Workspace carve (~140 MB total)
  char* w = (char*)d_ws;
  size_t off = 0;
  auto carve = [&](size_t bytes) -> void* {
    void* p = w + off; off += (bytes + 255) & ~(size_t)255; return p;
  };
  unsigned short* inpB  = (unsigned short*)carve((size_t)Bb * Tt * Hh * 2); // layer in / L1 out
  unsigned short* h1buf = (unsigned short*)carve((size_t)Bb * Tt * Hh * 2); // L0 out
  unsigned short* wfrag = (unsigned short*)carve((size_t)Ll * G4H * KC * 2);
  unsigned short* dfrag = (unsigned short*)carve((size_t)Vv * Hh * 2);
  float*          bsum  = (float*)         carve((size_t)Ll * G4H * 4);
  unsigned short* hsA   = (unsigned short*)carve((size_t)Bb * Hh * 2);
  unsigned short* hsB   = (unsigned short*)carve((size_t)Bb * Hh * 2);
  float*          cst   = (float*)         carve((size_t)Bb * Hh * 4);

  { long tot = (long)Ll * KT_G * NT_G * 32 * 16;
    prep_wcat<<<(unsigned)((tot + 255) / 256), 256, 0, stream>>>(Wih, Whh, wfrag); }
  { long tot = (long)KT_D * NT_D * 32 * 16;
    prep_dec<<<(unsigned)((tot + 255) / 256), 256, 0, stream>>>(decW, dfrag); }
  { int tot = Ll * G4H;
    prep_bias<<<(tot + 255) / 256, 256, 0, stream>>>(bih, bhh, bsum); }
  { long tot = (long)Bb * Tt * Hh;
    embed_kernel<<<(unsigned)((tot + 255) / 256), 256, 0, stream>>>(x, y, emb, yW, yb, inpB); }

  const unsigned short* lin = inpB;
  unsigned short* lout = h1buf;
  for (int l = 0; l < Ll; ++l) {
    { long tot = (long)Bb * Hh;
      init_state<<<(unsigned)((tot + 255) / 256), 256, 0, stream>>>(hsA, hsB, cst); }
    for (int t = 0; t < Tt; ++t) {
      const unsigned short* hp = (t & 1) ? hsB : hsA;
      unsigned short*       hn = (t & 1) ? hsA : hsB;
      lstm_step<<<dim3(Bb / 32, Hh / 64), 512, 0, stream>>>(
          lin, wfrag + (size_t)l * G4H * KC, bsum + (size_t)l * G4H,
          hp, hn, cst, lout, t);
    }
    const unsigned short* nlin = lout;
    lout = (l == 0) ? inpB : h1buf;   // layer1 writes back into inpB (free)
    lin  = nlin;
  }

  decode_gemm<<<dim3((Bb * Tt) / 64, Vv / 64), 256, 0, stream>>>(lin, dfrag, decb, out);
}
